// DETR3DCrossAtnn_78950088835361
// MI455X (gfx1250) — compile-verified, matrix operand order checked by reading, not run on hardware
//
#include <hip/hip_runtime.h>
#include <hip/hip_bf16.h>
#include <math.h>

typedef __attribute__((ext_vector_type(16))) _Float16 v16h;
typedef __attribute__((ext_vector_type(8)))  float    v8f;

#define B_  2
#define V_  6
#define Q_  900
#define E_  256
#define L_  4
#define M_  (B_ * Q_)   // 1800

// ---------------------------------------------------------------------------
// Block-wide sum over 256 threads (tree in shared memory).
// ---------------------------------------------------------------------------
__device__ __forceinline__ float block_sum_256(float v, float* red, int t) {
    red[t] = v;
    __syncthreads();
    #pragma unroll
    for (int s = 128; s > 0; s >>= 1) {
        if (t < s) red[t] += red[t + s];
        __syncthreads();
    }
    float r = red[0];
    __syncthreads();
    return r;
}

// ---------------------------------------------------------------------------
// Kernel 1: per (b,q) row:
//   - attention logits (256x24 dot) + sigmoid
//   - per-view projection -> normalized grid coords + mask
//   - PE layer 1: relu(layernorm(ref @ pe_w1 + pe_b1))
// ---------------------------------------------------------------------------
__global__ __launch_bounds__(256) void k_setup(
    const float* __restrict__ query,   // (B,Q,E)
    const float* __restrict__ refpos,  // (B,Q,3)
    const float* __restrict__ cam,     // (B,V,3,4)
    const float* __restrict__ img,     // (B,2) [H,W]
    const float* __restrict__ attn_w,  // (E, V*L)
    const float* __restrict__ attn_b,  // (V*L)
    const float* __restrict__ pe_w1,   // (3, E)
    const float* __restrict__ pe_b1,   // (E)
    const float* __restrict__ pe_g1,   // (E)
    const float* __restrict__ pe_be1,  // (E)
    float* __restrict__ attn_o,        // (M, 24)
    float* __restrict__ grid_o,        // (M, 6, 4) = gx, gy, mask, pad
    float* __restrict__ p1_o)          // (M, 256)
{
    const int row = blockIdx.x;        // 0..M-1
    const int b   = row / Q_;
    const int t   = threadIdx.x;       // 0..255
    __shared__ float red[256];

    // --- attention logits (threads 0..23) ---
    if (t < V_ * L_) {
        const float* qr = query + (size_t)row * E_;
        float s = attn_b[t];
        #pragma unroll 4
        for (int k = 0; k < E_; ++k) s += qr[k] * attn_w[k * (V_ * L_) + t];
        attn_o[(size_t)row * 24 + t] = 1.0f / (1.0f + expf(-s));
    }

    const float rx = refpos[row * 3 + 0];
    const float ry = refpos[row * 3 + 1];
    const float rz = refpos[row * 3 + 2];

    // --- projection per view (threads 0..5) ---
    if (t < V_) {
        const float* c = cam + (size_t)(b * V_ + t) * 12;
        float px = c[0] * rx + c[1] * ry + c[2]  * rz + c[3];
        float py = c[4] * rx + c[5] * ry + c[6]  * rz + c[7];
        float pz = c[8] * rx + c[9] * ry + c[10] * rz + c[11];
        const float eps = 1e-5f;
        float zc = fmaxf(pz, eps);
        float x = (px / zc) / img[b * 2 + 1];   // / img W
        float y = (py / zc) / img[b * 2 + 0];   // / img H
        bool m = (pz >= eps) && (x > 0.0f) && (x < 1.0f) && (y > 0.0f) && (y < 1.0f);
        float* g = grid_o + ((size_t)row * V_ + t) * 4;
        g[0] = 2.0f * x - 1.0f;
        g[1] = 2.0f * y - 1.0f;
        g[2] = m ? 1.0f : 0.0f;
        g[3] = 0.0f;
    }

    // --- PE layer 1 (all 256 threads, one output feature each) ---
    float h = pe_b1[t] + rx * pe_w1[t] + ry * pe_w1[E_ + t] + rz * pe_w1[2 * E_ + t];
    float mean = block_sum_256(h, red, t) * (1.0f / 256.0f);
    float d = h - mean;
    float var = block_sum_256(d * d, red, t) * (1.0f / 256.0f);
    float rs = rsqrtf(var + 1e-5f);
    float p = d * rs * pe_g1[t] + pe_be1[t];
    p1_o[(size_t)row * E_ + t] = fmaxf(p, 0.0f);
}

// ---------------------------------------------------------------------------
// Kernel 2: bilinear gather + mask + attention-weighted reduction over (l,v).
// One block per (b,q); thread = channel. Mask is uniform per block -> no
// per-lane divergence on the skip branch.
// ---------------------------------------------------------------------------
__global__ __launch_bounds__(256) void k_sample(
    const float* __restrict__ f0, const float* __restrict__ f1,
    const float* __restrict__ f2, const float* __restrict__ f3,
    const float* __restrict__ attn,   // (M,24)
    const float* __restrict__ grid,   // (M,6,4)
    float* __restrict__ acc_o)        // (M,256)
{
    const int rowq = blockIdx.x;      // b*Q + q
    const int b    = rowq / Q_;
    const int c    = threadIdx.x;     // channel

    __shared__ float sa[24];
    __shared__ float sg[24];
    if (c < 24) sa[c] = attn[(size_t)rowq * 24 + c];
    if (c < 24) sg[c] = grid[(size_t)rowq * 24 + c];
    __syncthreads();

    const float* feats[4] = { f0, f1, f2, f3 };
    const int HH[4] = { 116, 58, 29, 15 };
    const int WW[4] = { 200, 100, 50, 25 };

    float acc = 0.0f;
    #pragma unroll
    for (int l = 0; l < L_; ++l) {
        const int H = HH[l], W = WW[l];
        const float* fl = feats[l];
        #pragma unroll
        for (int v = 0; v < V_; ++v) {
            if (sg[v * 4 + 2] == 0.0f) continue;       // masked view
            const float wA = sa[l * V_ + v];
            const float x = ((sg[v * 4 + 0] + 1.0f) * (float)W - 1.0f) * 0.5f;
            const float y = ((sg[v * 4 + 1] + 1.0f) * (float)H - 1.0f) * 0.5f;
            const float x0 = floorf(x), y0 = floorf(y);
            const float wx = x - x0, wy = y - y0;
            const int ix = (int)x0, iy = (int)y0;
            const float* fp = fl + ((size_t)(b * V_ + v) * E_ + c) * (size_t)(H * W);
            float v00 = (ix >= 0     && ix < W     && iy >= 0     && iy < H)     ? fp[iy * W + ix]           : 0.0f;
            float v10 = (ix + 1 >= 0 && ix + 1 < W && iy >= 0     && iy < H)     ? fp[iy * W + ix + 1]       : 0.0f;
            float v01 = (ix >= 0     && ix < W     && iy + 1 >= 0 && iy + 1 < H) ? fp[(iy + 1) * W + ix]     : 0.0f;
            float v11 = (ix + 1 >= 0 && ix + 1 < W && iy + 1 >= 0 && iy + 1 < H) ? fp[(iy + 1) * W + ix + 1] : 0.0f;
            float bil = v00 * (1.0f - wx) * (1.0f - wy) + v10 * wx * (1.0f - wy)
                      + v01 * (1.0f - wx) * wy          + v11 * wx * wy;
            acc += wA * bil;
        }
    }
    acc_o[(size_t)rowq * E_ + c] = acc;
}

// ---------------------------------------------------------------------------
// Kernel 3a: convert a 256x256 f32 weight (row-major K x N) into f16 stored in
// WMMA-B-fragment order, so the GEMM can load each lane's fragment as one
// contiguous 32-byte vector:
//   Bf[((kt*16 + nt)*32 + lane)*16 + j]
// lane -> col = nt*16 + (lane&15), kg = lane>>4
// j    -> jj = j&7, khalf = j>>3 ; K = kt*32 + khalf*16 + kg*8 + jj
// (matches ISA §7.12.2 16-bit B-matrix 32x16 layout)
// ---------------------------------------------------------------------------
__global__ __launch_bounds__(256) void k_cvtw(
    const float* __restrict__ Wm,     // 256 x 256 f32
    _Float16* __restrict__ Bf)        // 65536 f16
{
    const int idx = blockIdx.x * 256 + threadIdx.x;   // 0..65535
    const int j     = idx & 15;
    const int lane  = (idx >> 4) & 31;
    const int nt    = (idx >> 9) & 15;
    const int kt    = idx >> 13;
    const int kg    = lane >> 4;
    const int jj    = j & 7;
    const int khalf = j >> 3;
    const int K     = kt * 32 + khalf * 16 + kg * 8 + jj;
    const int col   = nt * 16 + (lane & 15);
    Bf[idx] = (_Float16)Wm[K * 256 + col];
}

// ---------------------------------------------------------------------------
// Load one A fragment (16x32 f16, ISA §7.12.2) from a row-major f32 matrix,
// converting on the fly; rows >= M produce a zero fragment.
// ---------------------------------------------------------------------------
__device__ __forceinline__ v16h load_a_frag(const float* __restrict__ A,
                                            int row, int kk, int kg, int M) {
    v16h a;
    if (row < M) {
        const float* ap = A + (size_t)row * 256 + kk + kg * 8;
        #pragma unroll
        for (int j = 0; j < 8; ++j) {
            a[j]     = (_Float16)ap[j];        // K = kk + kg*8 + j
            a[8 + j] = (_Float16)ap[16 + j];   // K = kk + 16 + kg*8 + j
        }
    } else {
        #pragma unroll
        for (int j = 0; j < 16; ++j) a[j] = (_Float16)0.0f;
    }
    return a;
}

// ---------------------------------------------------------------------------
// Kernel 3b: C = A(Mx256,f32) @ W(256x256, pre-converted f16 fragments)
//            (+bias), via v_wmma_f32_16x16x32_f16.
// 512 threads = 16 waves; wave w owns N-tile [16w,16w+16) and TWO M-tiles
// (32 rows per block) sharing each B fragment: 16 static WMMAs after unroll.
// ---------------------------------------------------------------------------
__global__ __launch_bounds__(512) void k_gemm32(
    const float*    __restrict__ A,    // M x 256
    const _Float16* __restrict__ Bf,   // fragment-ordered f16 weights
    const float*    __restrict__ bias, // 256 or nullptr
    float*          __restrict__ Cout, // M x 256
    int M)
{
    const int wave = threadIdx.x >> 5;   // 0..15 = nt
    const int lane = threadIdx.x & 31;
    const int m0   = blockIdx.x * 32;
    const int n0   = wave * 16;
    const int lr   = lane & 15;
    const int kg   = lane >> 4;
    const int row0 = m0 + lr;
    const int row1 = m0 + 16 + lr;
    const int col  = n0 + lr;

    v8f c0 = {};
    v8f c1 = {};
    #pragma unroll
    for (int kt = 0; kt < 8; ++kt) {
        const int kk = kt * 32;
        v16h a0 = load_a_frag(A, row0, kk, kg, M);
        v16h a1 = load_a_frag(A, row1, kk, kg, M);
        const v16h bfrag = *(const v16h*)(Bf + ((size_t)(kt * 16 + wave) * 32 + lane) * 16);
        c0 = __builtin_amdgcn_wmma_f32_16x16x32_f16(false, a0, false, bfrag,
                                                    (short)0, c0, false, false);
        c1 = __builtin_amdgcn_wmma_f32_16x16x32_f16(false, a1, false, bfrag,
                                                    (short)0, c1, false, false);
    }

    const float bb = bias ? bias[col] : 0.0f;
    #pragma unroll
    for (int r = 0; r < 8; ++r) {
        const int orow0 = m0 + r + kg * 8;        // M = r + 8*(lane/16)
        if (orow0 < M)      Cout[(size_t)orow0 * 256 + col] = c0[r] + bb;
        const int orow1 = orow0 + 16;
        if (orow1 < M)      Cout[(size_t)orow1 * 256 + col] = c1[r] + bb;
    }
}

// ---------------------------------------------------------------------------
// Kernel 4: out = gout + relu(layernorm2(p2raw)) ; p2raw already has pe_b2.
// ---------------------------------------------------------------------------
__global__ __launch_bounds__(256) void k_finalize(
    const float* __restrict__ gout,   // (M,256), includes out_b
    const float* __restrict__ p2raw,  // (M,256), includes pe_b2
    const float* __restrict__ g2,
    const float* __restrict__ be2,
    float* __restrict__ out)          // (M,256)
{
    const int row = blockIdx.x;
    const int t   = threadIdx.x;
    __shared__ float red[256];

    float h = p2raw[(size_t)row * E_ + t];
    float mean = block_sum_256(h, red, t) * (1.0f / 256.0f);
    float d = h - mean;
    float var = block_sum_256(d * d, red, t) * (1.0f / 256.0f);
    float rs = rsqrtf(var + 1e-5f);
    float p = fmaxf(d * rs * g2[t] + be2[t], 0.0f);
    out[(size_t)row * E_ + t] = gout[(size_t)row * E_ + t] + p;
}

// ---------------------------------------------------------------------------
extern "C" void kernel_launch(void* const* d_in, const int* in_sizes, int n_in,
                              void* d_out, int out_size, void* d_ws, size_t ws_size,
                              hipStream_t stream) {
    const float* query   = (const float*)d_in[0];
    const float* f0      = (const float*)d_in[1];
    const float* f1      = (const float*)d_in[2];
    const float* f2      = (const float*)d_in[3];
    const float* f3      = (const float*)d_in[4];
    const float* refpos  = (const float*)d_in[5];
    const float* cam     = (const float*)d_in[6];
    const float* img     = (const float*)d_in[7];
    const float* attn_w  = (const float*)d_in[8];
    const float* attn_b  = (const float*)d_in[9];
    const float* out_w   = (const float*)d_in[10];
    const float* out_b   = (const float*)d_in[11];
    const float* pe_w1   = (const float*)d_in[12];
    const float* pe_b1   = (const float*)d_in[13];
    const float* pe_g1   = (const float*)d_in[14];
    const float* pe_be1  = (const float*)d_in[15];
    const float* pe_w2   = (const float*)d_in[16];
    const float* pe_b2   = (const float*)d_in[17];
    const float* pe_g2   = (const float*)d_in[18];
    const float* pe_be2  = (const float*)d_in[19];
    float* out = (float*)d_out;

    // Workspace layout (floats): attn | grid | p1 | acc | gout | p2raw | f16 weights
    float* ws      = (float*)d_ws;
    float* attn_ws = ws;                               // M*24
    float* grid_ws = attn_ws + (size_t)M_ * 24;        // M*24
    float* p1      = grid_ws + (size_t)M_ * 24;        // M*256
    float* acc     = p1      + (size_t)M_ * E_;        // M*256
    float* gout    = acc     + (size_t)M_ * E_;        // M*256
    float* p2r     = gout    + (size_t)M_ * E_;        // M*256
    _Float16* wf_out = (_Float16*)(p2r + (size_t)M_ * E_);  // 65536 f16 (32B aligned)
    _Float16* wf_pe  = wf_out + 65536;                      // 65536 f16

    // Pre-convert weights to fragment-ordered f16 (reused by both GEMMs).
    k_cvtw<<<256, 256, 0, stream>>>(out_w, wf_out);
    k_cvtw<<<256, 256, 0, stream>>>(pe_w2, wf_pe);

    k_setup<<<M_, 256, 0, stream>>>(query, refpos, cam, img, attn_w, attn_b,
                                    pe_w1, pe_b1, pe_g1, pe_be1,
                                    attn_ws, grid_ws, p1);

    k_sample<<<M_, 256, 0, stream>>>(f0, f1, f2, f3, attn_ws, grid_ws, acc);

    const int mtiles32 = (M_ + 31) / 32;   // 57
    k_gemm32<<<mtiles32, 512, 0, stream>>>(acc, wf_out, out_b, gout, M_);
    k_gemm32<<<mtiles32, 512, 0, stream>>>(p1, wf_pe, pe_b2, p2r, M_);

    k_finalize<<<M_, 256, 0, stream>>>(gout, p2r, pe_g2, pe_be2, out);
}
